// GraphEmbedder_11768210391176
// MI455X (gfx1250) — compile-verified, matrix operand order checked by reading, not run on hardware
//
#include <hip/hip_runtime.h>
#include <hip/hip_bf16.h>

// ---------------------------------------------------------------------------
// GCN forward for MI455X (gfx1250): dense projections via v_wmma_f32_16x16x32_bf16,
// sparse gather/scatter via float4 + global_atomic_add_f32 (one wave32 per edge).
// ---------------------------------------------------------------------------

#define DIM      128
#define NSLOPE   0.01f
#define LN_EPS   1e-5f
#define WT_STRIDE 138   // padded LDS row stride (halfwords); 138*2B is 4B-aligned,
                        // stride mod 64 words == 5 -> half-wave bank sets disjoint

typedef __attribute__((ext_vector_type(16))) __bf16 v16bf;
typedef __attribute__((ext_vector_type(8)))  float  v8f;

union ABFrag { v16bf v; unsigned int w[8]; };

__device__ __forceinline__ unsigned short f32_to_bf16(float f) {
  unsigned int u = __float_as_uint(f);
  u += 0x7FFFu + ((u >> 16) & 1u);          // round-to-nearest-even
  return (unsigned short)(u >> 16);
}
__device__ __forceinline__ unsigned int pack_bf16x2(float a, float b) {
  unsigned int ua = __float_as_uint(a);
  unsigned int ub = __float_as_uint(b);
  ua += 0x7FFFu + ((ua >> 16) & 1u);
  ub += 0x7FFFu + ((ub >> 16) & 1u);
  return (ua >> 16) | (ub & 0xFFFF0000u);
}
__device__ __forceinline__ float leaky(float v) {
  return v >= 0.0f ? v : NSLOPE * v;
}

// ---------------------------------------------------------------------------
// Out[M,128] = A[M,128] @ W[128,128] (+ bias).  256 threads = 8 waves;
// W staged bf16-transposed in LDS; one 16-row M-tile per wave.
// Per K-chunk: all 8 B fragments are loaded up front, then 8 WMMAs issue
// back-to-back (single dscnt wait, dense XDL issue, next chunk's loads overlap).
// ---------------------------------------------------------------------------
__global__ void __launch_bounds__(256)
gemm_wmma_bf16(const float* __restrict__ Amat, const float* __restrict__ Wmat,
               const float* __restrict__ bias, float* __restrict__ Out, int M) {
  __shared__ unsigned short Wt[DIM * WT_STRIDE];
  const int tid = threadIdx.x;
  for (int idx = tid; idx < DIM * DIM; idx += 256) {
    const int k = idx >> 7, n = idx & 127;
    Wt[n * WT_STRIDE + k] = f32_to_bf16(Wmat[idx]);   // Wt[n][k] = W[k][n]
  }
  __syncthreads();

  const int wave  = tid >> 5;
  const int lane  = tid & 31;
  const int mtile = blockIdx.x * 8 + wave;
  if (mtile * 16 >= M) return;                        // uniform per-wave guard
  const int hi = lane >> 4;                           // upper half-wave: K += 8
  const int lo = lane & 15;
  int row = mtile * 16 + lo;
  if (row >= M) row = M - 1;                          // safety (M%16==0 here)

  v8f zero = {};
  v8f acc[8];
#pragma unroll
  for (int i = 0; i < 8; ++i) acc[i] = zero;

  const float* ap = Amat + (size_t)row * DIM;
  const unsigned short* wbase = Wt + lo * WT_STRIDE;

#pragma unroll
  for (int kc = 0; kc < 4; ++kc) {
    const int kbase = kc * 32 + hi * 8;

    // A fragment: 16-bit A 16x32 layout (ISA 7.12.2)
    ABFrag a;
#pragma unroll
    for (int p = 0; p < 4; ++p) {
      float2 f0 = *(const float2*)(ap + kbase + 2 * p);        // VGPR p   : K=kbase+2p,+1
      float2 f1 = *(const float2*)(ap + kbase + 16 + 2 * p);   // VGPR 4+p : K=kbase+16+2p,+1
      a.w[p]     = pack_bf16x2(f0.x, f0.y);
      a.w[4 + p] = pack_bf16x2(f1.x, f1.y);
    }

    // Load ALL 8 B fragments for this K-chunk first (B mirrors A's K mapping
    // with N = lane%16), then run the 8 WMMAs as an uninterrupted chain.
    ABFrag b[8];
#pragma unroll
    for (int nt = 0; nt < 8; ++nt) {
      const unsigned short* wp = wbase + nt * (16 * WT_STRIDE);
#pragma unroll
      for (int p = 0; p < 4; ++p) {
        b[nt].w[p]     = *(const unsigned int*)(wp + kbase + 2 * p);
        b[nt].w[4 + p] = *(const unsigned int*)(wp + kbase + 16 + 2 * p);
      }
    }

#pragma unroll
    for (int nt = 0; nt < 8; ++nt) {
      acc[nt] = __builtin_amdgcn_wmma_f32_16x16x32_bf16(
          false, a.v, false, b[nt].v, (short)0, acc[nt], false, false);
    }
  }

  // C/D layout: lane%16 = N col, VGPR r -> M row r (+8 for upper half-wave)
#pragma unroll
  for (int nt = 0; nt < 8; ++nt) {
    const int col = nt * 16 + lo;
    const float bv = bias ? bias[col] : 0.0f;
#pragma unroll
    for (int r = 0; r < 8; ++r) {
      const int orow = mtile * 16 + r + hi * 8;
      if (orow < M) Out[(size_t)orow * DIM + col] = acc[nt][r] + bv;
    }
  }
}

// ---------------------------------------------------------------------------
// Graph-structure precompute
// ---------------------------------------------------------------------------
__global__ void deg_init_kernel(float* __restrict__ deg, int n) {
  int i = blockIdx.x * blockDim.x + threadIdx.x;
  if (i < n) deg[i] = 1.0f;                  // self-loop
}
__global__ void deg_acc_kernel(const int* __restrict__ dst, float* __restrict__ deg, int e) {
  int i = blockIdx.x * blockDim.x + threadIdx.x;
  if (i < e) atomicAdd(&deg[dst[i]], 1.0f);
}
__global__ void dis_kernel(float* __restrict__ deg, int n) {
  int i = blockIdx.x * blockDim.x + threadIdx.x;
  if (i < n) deg[i] = rsqrtf(deg[i]);        // deg >= 1 always
}
__global__ void norm_kernel(const int* __restrict__ src, const int* __restrict__ dst,
                            const float* __restrict__ dis, float* __restrict__ norm, int e) {
  int i = blockIdx.x * blockDim.x + threadIdx.x;
  if (i < e) norm[i] = dis[src[i]] * dis[dst[i]];
}

// out[i][d] = bias[d] + m[i][d] * dis[i]^2   (bias + self-loop message)
__global__ void conv_init_kernel(const float* __restrict__ m, const float* __restrict__ dis,
                                 const float* __restrict__ bias, float* __restrict__ out,
                                 long long total) {
  long long i = (long long)blockIdx.x * blockDim.x + threadIdx.x;
  if (i >= total) return;
  int node = (int)(i >> 7), d = (int)(i & 127);
  float s = dis[node];
  out[i] = bias[d] + m[i] * s * s;
}

// one wave32 per edge: lane carries 4 contiguous floats (512B/edge, coalesced)
__global__ void __launch_bounds__(256)
scatter_kernel(const float* __restrict__ m, float* __restrict__ out,
               const int* __restrict__ src, const int* __restrict__ dst,
               const float* __restrict__ norm, int e) {
  int edge = (int)((blockIdx.x * (long long)blockDim.x + threadIdx.x) >> 5);
  int lane = threadIdx.x & 31;
  if (edge >= e) return;
  const int   s  = src[edge];
  const int   d  = dst[edge];
  const float nm = norm[edge];
  const float4 v = ((const float4*)(m + (size_t)s * DIM))[lane];
  float* op = out + (size_t)d * DIM + lane * 4;
  atomicAdd(op + 0, v.x * nm);
  atomicAdd(op + 1, v.y * nm);
  atomicAdd(op + 2, v.z * nm);
  atomicAdd(op + 3, v.w * nm);
}

__global__ void leaky_kernel(float* __restrict__ p, long long total) {
  long long i = (long long)blockIdx.x * blockDim.x + threadIdx.x;
  if (i < total) p[i] = leaky(p[i]);
}

// h_out = leaky( LN(c + res) * gamma + beta ), one wave32 per node row
__global__ void __launch_bounds__(256)
ln_res_leaky_kernel(const float* __restrict__ c, const float* __restrict__ res,
                    const float* __restrict__ gamma, const float* __restrict__ beta,
                    float* __restrict__ out, int n) {
  int node = (int)((blockIdx.x * (long long)blockDim.x + threadIdx.x) >> 5);
  int lane = threadIdx.x & 31;
  if (node >= n) return;
  const float4 cv = ((const float4*)(c   + (size_t)node * DIM))[lane];
  const float4 rv = ((const float4*)(res + (size_t)node * DIM))[lane];
  float4 v = { cv.x + rv.x, cv.y + rv.y, cv.z + rv.z, cv.w + rv.w };
  float s  = v.x + v.y + v.z + v.w;
  float ss = v.x * v.x + v.y * v.y + v.z * v.z + v.w * v.w;
#pragma unroll
  for (int off = 16; off > 0; off >>= 1) {
    s  += __shfl_xor(s,  off, 32);
    ss += __shfl_xor(ss, off, 32);
  }
  const float mean = s * (1.0f / DIM);
  const float var  = ss * (1.0f / DIM) - mean * mean;
  const float rstd = rsqrtf(var + LN_EPS);
  const float4 g = ((const float4*)gamma)[lane];
  const float4 b = ((const float4*)beta)[lane];
  float4 y;
  y.x = leaky((v.x - mean) * rstd * g.x + b.x);
  y.y = leaky((v.y - mean) * rstd * g.y + b.y);
  y.z = leaky((v.z - mean) * rstd * g.z + b.z);
  y.w = leaky((v.w - mean) * rstd * g.w + b.w);
  ((float4*)(out + (size_t)node * DIM))[lane] = y;
}

__global__ void zero_kernel(float* __restrict__ p, int n) {
  int i = blockIdx.x * blockDim.x + threadIdx.x;
  if (i < n) p[i] = 0.0f;
}

__global__ void __launch_bounds__(256)
pool_scatter_kernel(const float* __restrict__ h, const int* __restrict__ batch,
                    float* __restrict__ sums, float* __restrict__ cnts, int n) {
  int node = (int)((blockIdx.x * (long long)blockDim.x + threadIdx.x) >> 5);
  int lane = threadIdx.x & 31;
  if (node >= n) return;
  const int g = batch[node];
  const float4 v = ((const float4*)(h + (size_t)node * DIM))[lane];
  float* sp = sums + (size_t)g * DIM + lane * 4;
  atomicAdd(sp + 0, v.x);
  atomicAdd(sp + 1, v.y);
  atomicAdd(sp + 2, v.z);
  atomicAdd(sp + 3, v.w);
  if (lane == 0) atomicAdd(&cnts[g], 1.0f);
}

__global__ void pool_div_kernel(float* __restrict__ sums, const float* __restrict__ cnts, int total) {
  int i = blockIdx.x * blockDim.x + threadIdx.x;
  if (i < total) sums[i] /= fmaxf(cnts[i >> 7], 1.0f);
}

// ---------------------------------------------------------------------------
extern "C" void kernel_launch(void* const* d_in, const int* in_sizes, int n_in,
                              void* d_out, int out_size, void* d_ws, size_t ws_size,
                              hipStream_t stream) {
  const float* x        = (const float*)d_in[0];
  /* d_in[1] edge_attr: unused by reference */
  const float* W_in     = (const float*)d_in[2];
  const float* b_in     = (const float*)d_in[3];
  const float* Ws       = (const float*)d_in[4];
  const float* bs       = (const float*)d_in[5];
  const float* gammas   = (const float*)d_in[6];
  const float* betas    = (const float*)d_in[7];
  const float* W_out    = (const float*)d_in[8];
  const float* b_out    = (const float*)d_in[9];
  const int*   eidx     = (const int*)d_in[10];   // [2,E] (integer -> int32 per harness)
  const int*   batch    = (const int*)d_in[11];

  const int N = in_sizes[0] / DIM;
  const int E = in_sizes[10] / 2;
  const int G = out_size / DIM;
  const int L = in_sizes[4] / (DIM * DIM);
  const long long ND = (long long)N * DIM;

  const int* src = eidx;
  const int* dst = eidx + E;

  // ---- workspace carve-up -------------------------------------------------
  char* ws = (char*)d_ws;
  auto alloc = [&](size_t bytes) -> float* {
    float* p = (float*)ws;
    ws += (bytes + 255) & ~(size_t)255;
    return p;
  };
  float* dis  = alloc((size_t)N * 4);          // degree -> rsqrt(degree), in place
  float* nrm  = alloc((size_t)E * 4);
  float* bufA = alloc((size_t)ND * 4);
  float* bufB = alloc((size_t)ND * 4);
  float* bufC = alloc((size_t)ND * 4);
  float* sums = alloc(((size_t)G * DIM + G) * 4);
  float* cnts = sums + (size_t)G * DIM;
  (void)ws_size; (void)n_in;

  const int T = 256;
  const int gN   = (N + T - 1) / T;
  const int gE   = (E + T - 1) / T;
  const int gND  = (int)((ND + T - 1) / T);
  const int gEW  = (E + 7) / 8;                // one wave per edge, 8 waves/block
  const int gNW  = (N + 7) / 8;                // one wave per node
  const int gMM  = (((N + 15) / 16) + 7) / 8;  // gemm blocks over node rows

  // ---- graph structure ----------------------------------------------------
  deg_init_kernel<<<gN, T, 0, stream>>>(dis, N);
  deg_acc_kernel <<<gE, T, 0, stream>>>(dst, dis, E);
  dis_kernel     <<<gN, T, 0, stream>>>(dis, N);
  norm_kernel    <<<gE, T, 0, stream>>>(src, dst, dis, nrm, E);

  // ---- input layer: h = leaky(gcn_conv(x, W_in, b_in)) --------------------
  // m=bufA, conv out=bufB
  gemm_wmma_bf16  <<<gMM, T, 0, stream>>>(x, W_in, nullptr, bufA, N);
  conv_init_kernel<<<gND, T, 0, stream>>>(bufA, dis, b_in, bufB, ND);
  scatter_kernel  <<<gEW, T, 0, stream>>>(bufA, bufB, src, dst, nrm, E);
  leaky_kernel    <<<gND, T, 0, stream>>>(bufB, ND);

  // ---- hidden layers ------------------------------------------------------
  float* h = bufB;   // current features
  float* m = bufA;   // projection scratch (free after scatter)
  float* c = bufC;   // conv output
  for (int l = 0; l < L; ++l) {
    const float* Wl = Ws + (size_t)l * DIM * DIM;
    const float* bl = bs + (size_t)l * DIM;
    const float* gl = gammas + (size_t)l * DIM;
    const float* tl = betas + (size_t)l * DIM;
    gemm_wmma_bf16     <<<gMM, T, 0, stream>>>(h, Wl, nullptr, m, N);
    conv_init_kernel   <<<gND, T, 0, stream>>>(m, dis, bl, c, ND);
    scatter_kernel     <<<gEW, T, 0, stream>>>(m, c, src, dst, nrm, E);
    ln_res_leaky_kernel<<<gNW, T, 0, stream>>>(c, h, gl, tl, m, N);  // new h into m
    float* nh = m; m = h; h = nh;                                    // rotate
  }

  // ---- global mean pool + output projection -------------------------------
  const int poolTot = G * DIM + G;
  zero_kernel        <<<(poolTot + T - 1) / T, T, 0, stream>>>(sums, poolTot);
  pool_scatter_kernel<<<gNW, T, 0, stream>>>(h, batch, sums, cnts, N);
  pool_div_kernel    <<<(G * DIM + T - 1) / T, T, 0, stream>>>(sums, cnts, G * DIM);
  gemm_wmma_bf16     <<<(((G + 15) / 16) + 7) / 8, T, 0, stream>>>(
      sums, W_out, b_out, (float*)d_out, G);
}
